// ResidualVectorQuantizer_73117523247503
// MI455X (gfx1250) — matrix operand order repaired and checked by reading, not compiled, not run on hardware
//
#include <hip/hip_runtime.h>

typedef __attribute__((ext_vector_type(16))) __bf16 v16bf;
typedef __attribute__((ext_vector_type(8)))  float  v8f;
typedef __attribute__((ext_vector_type(4)))  unsigned int u32x4;
typedef __attribute__((ext_vector_type(8)))  int   i32x8;
typedef __attribute__((ext_vector_type(4)))  int   i32x4;

#define N_ROWS 131072
#define DIM    128
#define KCB    1024
#define QSTAGE 4

// ---------------------------------------------------------------------------
// Prep 1: split codebooks into bf16 hi/lo blob in WMMA B-fragment order.
// Blob layout per codeword (256 bf16 = 512B):
//   [chunk c:0..3][half h:0..1][part p:hi,lo][e:0..15], k = 32c + 8h + (e&7) + 16*(e>=8)
// ---------------------------------------------------------------------------
__global__ __launch_bounds__(256) void rvq_prep_blob(const float* __restrict__ cb,
                                                     __bf16* __restrict__ blob) {
    int g = blockIdx.x * 256 + threadIdx.x;      // 0 .. Q*K*D-1 (524288)
    float v = cb[g];
    __bf16 hi = (__bf16)v;
    __bf16 lo = (__bf16)(v - (float)hi);
    int k  = g & (DIM - 1);
    int cw = g >> 7;                             // q*K + n
    int c  = k >> 5, kk = k & 31, h = (kk >> 3) & 1, e = (kk & 7) | ((kk >> 4) << 3);
    __bf16* base = blob + ((size_t)cw << 8) + c * 64 + h * 32 + e;
    base[0]  = hi;                               // p = 0
    base[16] = lo;                               // p = 1
}

// Prep 2: ||E||^2 per codeword (Q*K = 4096 values)
__global__ __launch_bounds__(256) void rvq_prep_e2(const float* __restrict__ cb,
                                                   float* __restrict__ e2) {
    int cw = blockIdx.x * 256 + threadIdx.x;     // 0..4095
    const float* p = cb + (size_t)cw * DIM;
    float s = 0.f;
#pragma unroll 8
    for (int k = 0; k < DIM; ++k) s += p[k] * p[k];
    e2[cw] = s;
}

__device__ __forceinline__ void ld4(const float* p, float* d) {
    float4 t = *(const float4*)p;
    d[0] = t.x; d[1] = t.y; d[2] = t.z; d[3] = t.w;
}

// ---------------------------------------------------------------------------
// TDM: DMA one 64-codeword group (32KB dense) into LDS, hardware-padding
// +4 dwords after every 128 dwords -> 528B stride per codeword in LDS.
// D# per CDNA5 ISA. 1-D tile: 8192 dwords.
// ---------------------------------------------------------------------------
__device__ __forceinline__ void tdm_load_group(const uint4* __restrict__ blobq,
                                               int g, unsigned lds_base, int buf) {
    unsigned long long ga =
        (unsigned long long)(const void*)(blobq + (size_t)g * 64 * 32);
    unsigned lds = lds_base + (buf ? 33792u : 0u);

    u32x4 g0;
    g0[0] = 1u;                                        // count=1, user mode
    g0[1] = lds;                                       // lds_addr (bytes)
    g0[2] = (unsigned)(ga & 0xFFFFFFFFu);              // global_addr[31:0]
    g0[3] = (unsigned)((ga >> 32) & 0x1FFFFFFu)        // global_addr[56:32]
          | 0x80000000u;                               // type=2 ("image")

    i32x8 g1;
    g1[0] = (int)((2u << 16)      // data_size = 4 bytes
                | (1u << 20)      // pad_enable
                | (6u << 22)      // pad_interval: 128 dwords
                | (3u << 25));    // pad_amount:   4 dwords
    g1[1] = (int)(8192u << 16);   // tensor_dim0[15:0] in [31:16]
    g1[2] = (int)(1u << 16);      // tensor_dim0[31:16]=0 | tensor_dim1[15:0]=1
    g1[3] = (int)(8192u << 16);   // tensor_dim1[31:16]=0 | tile_dim0=8192
    g1[4] = 0;                    // tile_dim1=0, tile_dim2=0 (unused)
    g1[5] = 8192;                 // tensor_dim0_stride[31:0]
    g1[6] = 0;
    g1[7] = 0;

    i32x4 z4 = {0, 0, 0, 0};
    i32x8 z8 = {0, 0, 0, 0, 0, 0, 0, 0};
    __builtin_amdgcn_tensor_load_to_lds(g0, g1, z4, z4, z8, 0);
}

// ---------------------------------------------------------------------------
// Main fused kernel: 256 threads = 8 waves, 128 rows/block.
// Residual lives in registers in the A-fragment layout:
//   row = w*16 + (lane&15); r[c][e] holds k = 32c + 8*(lane>>4) + (e&7) + 16*(e>=8)
// Codebook groups double-buffered in LDS via TDM, overlapped with WMMA.
// ---------------------------------------------------------------------------
__global__ __launch_bounds__(256) void rvq_main(const float* __restrict__ x,
                                                const float* __restrict__ cb,
                                                const uint4* __restrict__ blob,
                                                const float* __restrict__ e2,
                                                float* __restrict__ xq,
                                                int*   __restrict__ idx_out,
                                                float* __restrict__ loss_part) {
    __shared__ uint4 sB[2][64 * 33];  // 2 buffers: 64 cw * 132 dwords (528B stride)
    __shared__ int   sIdx[128];
    __shared__ float sRed[8];

    const int tid  = threadIdx.x;
    const int lane = tid & 31;
    const int w    = tid >> 5;
    const int half = lane >> 4;
    const int l15  = lane & 15;
    const int rowL = w * 16 + l15;
    const int rowG = blockIdx.x * 128 + rowL;
    const unsigned ldsBase = (unsigned)(size_t)(void*)(&sB[0][0]);

    float r[4][16];
    const float* xr = x + (size_t)rowG * DIM;
#pragma unroll
    for (int c = 0; c < 4; ++c) {
        int k0 = 32 * c + 8 * half;
        ld4(xr + k0,      &r[c][0]);
        ld4(xr + k0 + 4,  &r[c][4]);
        ld4(xr + k0 + 16, &r[c][8]);
        ld4(xr + k0 + 20, &r[c][12]);
    }

    float lossAcc = 0.f;
    const unsigned int* sBd = (const unsigned int*)&sB[0][0];

    for (int q = 0; q < QSTAGE; ++q) {
        // Build bf16 hi/lo A fragments from fp32 residual (once per stage).
        v16bf ah[4], al[4];
#pragma unroll
        for (int c = 0; c < 4; ++c)
#pragma unroll
            for (int e = 0; e < 16; ++e) {
                float v = r[c][e];
                __bf16 hv = (__bf16)v;
                ah[c][e] = hv;
                al[c][e] = (__bf16)(v - (float)hv);
            }

        float bv[8]; int bi[8];
#pragma unroll
        for (int i = 0; i < 8; ++i) { bv[i] = 3.4e38f; bi[i] = 0; }

        const uint4* blobq = blob + (size_t)q * KCB * 32;   // 32 uint4 per cw
        const float* e2q   = e2 + q * KCB;

        if (w == 0) tdm_load_group(blobq, 0, ldsBase, 0);   // prime buffer 0

        for (int g = 0; g < 16; ++g) {           // 16 groups of 64 codewords
            if (w == 0) {
                if (g < 15) {                    // prefetch next group, wait current
                    tdm_load_group(blobq, g + 1, ldsBase, (g + 1) & 1);
                    __builtin_amdgcn_s_wait_tensorcnt(1);
                } else {
                    __builtin_amdgcn_s_wait_tensorcnt(0);
                }
            }
            __syncthreads();                     // group g visible to all waves
            const int bufBase = (g & 1) ? 8448 : 0;   // dwords
#pragma unroll
            for (int tl = 0; tl < 4; ++tl) {     // 4 tiles of 16 codewords
                int n = (g << 6) + (tl << 4) + l15;
                v16bf bh[4], bl[4];
#pragma unroll
                for (int c = 0; c < 4; ++c) {
                    int dw = bufBase + ((tl << 4) + l15) * 132 + c * 32 + half * 16;
                    union { uint4 u[2]; v16bf v; } uh, ul;
                    uh.u[0] = *(const uint4*)(sBd + dw);
                    uh.u[1] = *(const uint4*)(sBd + dw + 4);
                    ul.u[0] = *(const uint4*)(sBd + dw + 8);
                    ul.u[1] = *(const uint4*)(sBd + dw + 12);
                    bh[c] = uh.v; bl[c] = ul.v;
                }
                v8f acc = {0.f,0.f,0.f,0.f,0.f,0.f,0.f,0.f};
#pragma unroll
                for (int c = 0; c < 4; ++c) {    // bf16x3: hi*hi + hi*lo + lo*hi
                    acc = __builtin_amdgcn_wmma_f32_16x16x32_bf16(false, ah[c], false, bh[c], (short)0, acc, false, false);
                    acc = __builtin_amdgcn_wmma_f32_16x16x32_bf16(false, ah[c], false, bl[c], (short)0, acc, false, false);
                    acc = __builtin_amdgcn_wmma_f32_16x16x32_bf16(false, al[c], false, bh[c], (short)0, acc, false, false);
                }
                float ee = e2q[n];
#pragma unroll
                for (int i = 0; i < 8; ++i) {    // score = ||E||^2 - 2 r.E (||r||^2 const)
                    float s = ee - 2.0f * acc[i];
                    if (s < bv[i]) { bv[i] = s; bi[i] = n; }
                }
            }
            __syncthreads();                     // done reading buf[g&1]
        }

        // argmin across the 16 lanes of each half (rows half*8 + i), first-index ties
        const int offs[4] = {1, 2, 4, 8};
#pragma unroll
        for (int o = 0; o < 4; ++o) {
            int off = offs[o];
#pragma unroll
            for (int i = 0; i < 8; ++i) {
                float ov = __shfl_xor(bv[i], off, 32);
                int   oi = __shfl_xor(bi[i], off, 32);
                if (ov < bv[i] || (ov == bv[i] && oi < bi[i])) { bv[i] = ov; bi[i] = oi; }
            }
        }
        if (l15 == 0) {
#pragma unroll
            for (int i = 0; i < 8; ++i) {
                int rl = w * 16 + half * 8 + i;
                sIdx[rl] = bi[i];
                idx_out[(size_t)(blockIdx.x * 128 + rl) * QSTAGE + q] = bi[i];
            }
        }
        __syncthreads();

        // gather selected codeword (fp32, L2-hot) and update residual; loss = sum(new^2)
        int sel = sIdx[rowL];
        const float* cwp = cb + ((size_t)q * KCB + sel) * DIM;
#pragma unroll
        for (int c = 0; c < 4; ++c) {
            int k0 = 32 * c + 8 * half;
            float qv[16];
            ld4(cwp + k0,      &qv[0]);
            ld4(cwp + k0 + 4,  &qv[4]);
            ld4(cwp + k0 + 16, &qv[8]);
            ld4(cwp + k0 + 20, &qv[12]);
#pragma unroll
            for (int e = 0; e < 16; ++e) {
                float nv = r[c][e] - qv[e];
                r[c][e] = nv;
                lossAcc += nv * nv;
            }
        }
    }

    // x_q = x - residual_final
    float* oq = xq + (size_t)rowG * DIM;
#pragma unroll
    for (int c = 0; c < 4; ++c) {
        int k0 = 32 * c + 8 * half;
#pragma unroll
        for (int j = 0; j < 4; ++j) {
            float4 t = *(const float4*)(xr + k0 + ((j & 1) ? 4 : 0) + ((j >> 1) ? 16 : 0));
            int eb = j * 4;
            float4 o = make_float4(t.x - r[c][eb], t.y - r[c][eb + 1],
                                   t.z - r[c][eb + 2], t.w - r[c][eb + 3]);
            *(float4*)(oq + k0 + ((j & 1) ? 4 : 0) + ((j >> 1) ? 16 : 0)) = o;
        }
    }

    // deterministic per-block loss partial
#pragma unroll
    for (int off = 16; off >= 1; off >>= 1) lossAcc += __shfl_xor(lossAcc, off, 32);
    if (lane == 0) sRed[w] = lossAcc;
    __syncthreads();
    if (tid == 0) {
        float s = 0.f;
#pragma unroll
        for (int i = 0; i < 8; ++i) s += sRed[i];
        loss_part[blockIdx.x] = s;
    }
}

// Deterministic final reduction: mean_loss = (1+beta)/(Q*N*D) * sum(residual^2)
__global__ __launch_bounds__(256) void rvq_reduce(const float* __restrict__ parts,
                                                  float* __restrict__ out_scalar) {
    __shared__ float s[256];
    int tid = threadIdx.x;
    float v = 0.f;
#pragma unroll
    for (int i = 0; i < 4; ++i) v += parts[tid + 256 * i];
    s[tid] = v;
    __syncthreads();
#pragma unroll
    for (int stride = 128; stride >= 1; stride >>= 1) {
        if (tid < stride) s[tid] += s[tid + stride];
        __syncthreads();
    }
    if (tid == 0)
        *out_scalar = s[0] * (1.25f / ((float)QSTAGE * (float)N_ROWS * (float)DIM));
}

extern "C" void kernel_launch(void* const* d_in, const int* in_sizes, int n_in,
                              void* d_out, int out_size, void* d_ws, size_t ws_size,
                              hipStream_t stream) {
    const float* x  = (const float*)d_in[0];   // [N, D]
    const float* cb = (const float*)d_in[1];   // [Q, K, D]

    float* fout = (float*)d_out;
    float* xq   = fout;                               // N*D floats
    float* loss = fout + (size_t)N_ROWS * DIM;        // 1 float
    int*   idxo = (int*)(fout + (size_t)N_ROWS * DIM + 1); // N*Q int32

    char*   ws    = (char*)d_ws;
    __bf16* blob  = (__bf16*)ws;                              // 2 MB
    float*  e2    = (float*)(ws + (size_t)2 * 1024 * 1024);   // 16 KB
    float*  parts = (float*)(ws + (size_t)2 * 1024 * 1024 + 16 * 1024); // 4 KB

    rvq_prep_blob<<<(QSTAGE * KCB * DIM) / 256, 256, 0, stream>>>(cb, blob);
    rvq_prep_e2<<<(QSTAGE * KCB) / 256, 256, 0, stream>>>(cb, e2);
    rvq_main<<<N_ROWS / 128, 256, 0, stream>>>(x, cb, (const uint4*)blob, e2,
                                               xq, idxo, parts);
    rvq_reduce<<<1, 256, 0, stream>>>(parts, loss);
}
